// ASAECF_85435489452125
// MI455X (gfx1250) — matrix operand order repaired and checked by reading, not compiled, not run on hardware
//
#include <hip/hip_runtime.h>
#include <hip/hip_bf16.h>
#include <stdint.h>

typedef __attribute__((ext_vector_type(16))) __bf16 v16bf;
typedef __attribute__((ext_vector_type(8)))  float  v8f;

#define ROWD 2128   // raw row length (K of GEMM1)
#define K2   2176   // padded K (34 tiles of 64)
#define HD   512    // hidden (N of GEMM1, K of GEMM2)
#define LD   256    // latent (N of GEMM2)
#define BD   8192   // batch
#define APITCH 72   // LDS A-tile pitch (64 + 8 pad, multiple of 8)

union FragU { v16bf v; uint4 q[2]; };

__device__ __forceinline__ v8f zero8() {
  v8f z = {0.f,0.f,0.f,0.f,0.f,0.f,0.f,0.f};
  return z;
}

__device__ __forceinline__ v8f wmma_bf16(v16bf a, v16bf b, v8f c) {
  return __builtin_amdgcn_wmma_f32_16x16x32_bf16(false, a, false, b, (short)0, c,
                                                 false, false);
}

// A-matrix 16x32 bf16 fragment: lanes 0-15 hold M=0..15 with
// K = col+0..15; lanes 16-31 hold K = col+16..31 (matches ldB).
__device__ __forceinline__ v16bf ldA(const __bf16* base, int pitch, int row,
                                     int col, int lane) {
  const __bf16* p = base + (size_t)(row + (lane & 15)) * pitch + col +
                    ((lane >> 4) << 4);
  FragU f;
  f.q[0] = *(const uint4*)(p);
  f.q[1] = *(const uint4*)(p + 8);
  return f.v;
}

// B-matrix 32x16 bf16 fragment: lane = N; lanes 0-15 hold K = col+0..15,
// lanes 16-31 hold K = col+16..31 (16 contiguous K per half).
__device__ __forceinline__ v16bf ldB(const __bf16* base, int pitch, int row,
                                     int col, int lane) {
  const __bf16* p = base + (size_t)(row + (lane & 15)) * pitch + col +
                    ((lane >> 4) << 4);
  FragU f;
  f.q[0] = *(const uint4*)(p);
  f.q[1] = *(const uint4*)(p + 8);
  return f.v;
}

__device__ __forceinline__ void split_bf16(float f, __bf16& hi, __bf16& lo) {
  hi = (__bf16)f;
  lo = (__bf16)(f - (float)hi);
}

// ---- Prep: W1 (ROWDxHD, row-major) -> permuted, transposed [HD][K2] hi/lo ----
// concat([xr, feat]) @ W1 == raw_row @ W1p  with  W1p[j] = W1[g(j)],
// g(j) = (j >= 128) ? j-128 : j+2000. K padded with zeros to K2.
__global__ void __launch_bounds__(256) prep_w1_kernel(
    const float* __restrict__ W1, __bf16* __restrict__ hi,
    __bf16* __restrict__ lo) {
  int id = blockIdx.x * 256 + threadIdx.x;   // total = K2*HD
  int n = id & (HD - 1);
  int k = id >> 9;
  if (k >= K2) return;
  float val = 0.f;
  if (k < ROWD) {
    int g = (k >= 128) ? (k - 128) : (k + 2000);
    val = W1[(size_t)g * HD + n];
  }
  __bf16 h, l;
  split_bf16(val, h, l);
  hi[(size_t)n * K2 + k] = h;
  lo[(size_t)n * K2 + k] = l;
}

// ---- Prep: W2 (HDxLD, row-major) -> transposed [LD][HD] hi/lo ----
__global__ void __launch_bounds__(256) prep_w2_kernel(
    const float* __restrict__ W2, __bf16* __restrict__ hi,
    __bf16* __restrict__ lo) {
  int id = blockIdx.x * 256 + threadIdx.x;   // total = HD*LD
  int n = id & (LD - 1);
  int k = id >> 8;
  if (k >= HD) return;
  float val = W2[(size_t)k * LD + n];
  __bf16 h, l;
  split_bf16(val, h, l);
  hi[(size_t)n * HD + k] = h;
  lo[(size_t)n * HD + k] = l;
}

// ---- GEMM1 helpers: register-carried A-tile pipeline ----
__device__ __forceinline__ void load_tile(float4* regs,
                                          const float* __restrict__ lookup,
                                          const int* rowIdx, int kb, int tid) {
#pragma unroll
  for (int i = 0; i < 8; i++) {
    int lin = i * 256 + tid;        // 0..2047
    int r = lin >> 4;               // row 0..127
    int c4 = (lin & 15) << 2;       // 0..60
    int gcol = kb + c4;
    if (gcol < ROWD) {
      const float* p = lookup + (size_t)rowIdx[r] * ROWD + gcol;
      regs[i] = *(const float4*)p;
      // prefetch one k-tile further ahead (1 per 128B)
      if ((lin & 7) == 0 && gcol + 64 < ROWD) __builtin_prefetch(p + 64, 0, 1);
    } else {
      regs[i] = make_float4(0.f, 0.f, 0.f, 0.f);
    }
  }
}

__device__ __forceinline__ void store_tile(const float4* regs, __bf16* sHi,
                                           __bf16* sLo, int tid) {
#pragma unroll
  for (int i = 0; i < 8; i++) {
    int lin = i * 256 + tid;
    int r = lin >> 4;
    int c4 = (lin & 15) << 2;
    float fv[4] = {regs[i].x, regs[i].y, regs[i].z, regs[i].w};
    union { __bf16 h[4]; uint2 u; } ph, pl;
#pragma unroll
    for (int e = 0; e < 4; e++) split_bf16(fv[e], ph.h[e], pl.h[e]);
    *(uint2*)(sHi + r * APITCH + c4) = ph.u;   // ds_store_b64
    *(uint2*)(sLo + r * APITCH + c4) = pl.u;
  }
}

// ---- GEMM1: h = relu(gather(rows) @ W1p + b1), h stored bf16 hi/lo planes ----
// grid = 2 towers x 64 m_tiles x 4 n_tiles = 512 blocks,
// block = 256 (8 waves: 4 m-groups x 2 n-groups)
__global__ void __launch_bounds__(256) gemm1_kernel(
    const long long* __restrict__ x,
    const float* __restrict__ userL, const float* __restrict__ itemL,
    const __bf16* __restrict__ w1uHi, const __bf16* __restrict__ w1uLo,
    const __bf16* __restrict__ w1iHi, const __bf16* __restrict__ w1iLo,
    const float* __restrict__ ub1, const float* __restrict__ ib1,
    __bf16* __restrict__ hUHi, __bf16* __restrict__ hULo,
    __bf16* __restrict__ hIHi, __bf16* __restrict__ hILo) {
  __shared__ __bf16 sA[2][128 * APITCH];   // hi plane, lo plane
  __shared__ int rowIdx[128];

  const int tid = threadIdx.x;
  const int lane = tid & 31;
  const int w = tid >> 5;
  const int wave_m = w & 3;    // 4 x 32 rows
  const int wave_n = w >> 2;   // 2 x 64 cols

  const int tower = blockIdx.x >> 8;
  const int rem = blockIdx.x & 255;
  const int m_tile = rem >> 2;
  const int n_tile = rem & 3;
  const int m0 = m_tile * 128;
  const int n0 = n_tile * 128;

  const float* lookup = tower ? itemL : userL;
  const __bf16* w1hi = tower ? w1iHi : w1uHi;
  const __bf16* w1lo = tower ? w1iLo : w1uLo;
  const float* b1 = tower ? ib1 : ub1;
  __bf16* hHi = tower ? hIHi : hUHi;
  __bf16* hLo = tower ? hILo : hULo;

  if (tid < 128) rowIdx[tid] = (int)x[(size_t)(m0 + tid) * 2 + tower];

  v8f acc[2][4];
#pragma unroll
  for (int i = 0; i < 2; i++)
#pragma unroll
    for (int j = 0; j < 4; j++) acc[i][j] = zero8();

  __syncthreads();
  float4 regs[8];
  load_tile(regs, lookup, rowIdx, 0, tid);

  for (int kb = 0; kb < K2; kb += 64) {
    __syncthreads();                       // all waves done reading sA
    store_tile(regs, &sA[0][0], &sA[1][0], tid);
    if (kb + 64 < K2) load_tile(regs, lookup, rowIdx, kb + 64, tid);
    __syncthreads();                       // sA tile ready

#pragma unroll
    for (int kk = 0; kk < 64; kk += 32) {
      v16bf aHi[2], aLo[2];
#pragma unroll
      for (int im = 0; im < 2; im++) {
        aHi[im] = ldA(&sA[0][0], APITCH, wave_m * 32 + im * 16, kk, lane);
        aLo[im] = ldA(&sA[1][0], APITCH, wave_m * 32 + im * 16, kk, lane);
      }
#pragma unroll
      for (int in_ = 0; in_ < 4; in_++) {
        int nrow = n0 + wave_n * 64 + in_ * 16;
        v16bf bHi = ldB(w1hi, K2, nrow, kb + kk, lane);
        v16bf bLo = ldB(w1lo, K2, nrow, kb + kk, lane);
#pragma unroll
        for (int im = 0; im < 2; im++) {
          acc[im][in_] = wmma_bf16(aHi[im], bHi, acc[im][in_]);  // hi*hi
          acc[im][in_] = wmma_bf16(aHi[im], bLo, acc[im][in_]);  // hi*lo
          acc[im][in_] = wmma_bf16(aLo[im], bHi, acc[im][in_]);  // lo*hi
        }
      }
    }
  }

  // epilogue: bias + relu, split to bf16 hi/lo, store h planes [BD][HD]
#pragma unroll
  for (int im = 0; im < 2; im++) {
    int mbase = m0 + wave_m * 32 + im * 16 + ((lane >> 4) << 3);
#pragma unroll
    for (int in_ = 0; in_ < 4; in_++) {
      int n = n0 + wave_n * 64 + in_ * 16 + (lane & 15);
      float bias = b1[n];
#pragma unroll
      for (int j = 0; j < 8; j++) {
        float hval = fmaxf(acc[im][in_][j] + bias, 0.f);
        __bf16 hh, hl;
        split_bf16(hval, hh, hl);
        size_t off = (size_t)(mbase + j) * HD + n;
        hHi[off] = hh;
        hLo[off] = hl;
      }
    }
  }
}

// ---- GEMM2 + dot: u = hU@W2u + b2u ; v = hI@W2i + b2i ; out = rowdot(u,v) ----
// grid = 128 (64 rows each), block = 256 (8 waves: 2 m-groups x 4 n-groups)
__global__ void __launch_bounds__(256) gemm2_dot_kernel(
    const __bf16* __restrict__ hUHi, const __bf16* __restrict__ hULo,
    const __bf16* __restrict__ hIHi, const __bf16* __restrict__ hILo,
    const __bf16* __restrict__ w2uHi, const __bf16* __restrict__ w2uLo,
    const __bf16* __restrict__ w2iHi, const __bf16* __restrict__ w2iLo,
    const float* __restrict__ b2u, const float* __restrict__ b2i,
    float* __restrict__ out) {
  __shared__ float dotLds[64];
  const int tid = threadIdx.x;
  const int lane = tid & 31;
  const int w = tid >> 5;
  const int wave_m = w & 1;    // 2 x 32 rows
  const int wave_n = w >> 1;   // 4 x 64 cols
  const int m0 = blockIdx.x * 64;

  if (tid < 64) dotLds[tid] = 0.f;
  __syncthreads();

  v8f uAcc[2][4], vAcc[2][4];
#pragma unroll
  for (int i = 0; i < 2; i++)
#pragma unroll
    for (int j = 0; j < 4; j++) { uAcc[i][j] = zero8(); vAcc[i][j] = zero8(); }

  for (int kk = 0; kk < HD; kk += 32) {
#pragma unroll
    for (int im = 0; im < 2; im++) {
      int mrow = m0 + wave_m * 32 + im * 16;
      v16bf aUHi = ldA(hUHi, HD, mrow, kk, lane);
      v16bf aULo = ldA(hULo, HD, mrow, kk, lane);
      v16bf aIHi = ldA(hIHi, HD, mrow, kk, lane);
      v16bf aILo = ldA(hILo, HD, mrow, kk, lane);
#pragma unroll
      for (int in_ = 0; in_ < 4; in_++) {
        int nrow = wave_n * 64 + in_ * 16;
        v16bf bUHi = ldB(w2uHi, HD, nrow, kk, lane);
        v16bf bULo = ldB(w2uLo, HD, nrow, kk, lane);
        uAcc[im][in_] = wmma_bf16(aUHi, bUHi, uAcc[im][in_]);
        uAcc[im][in_] = wmma_bf16(aUHi, bULo, uAcc[im][in_]);
        uAcc[im][in_] = wmma_bf16(aULo, bUHi, uAcc[im][in_]);
        v16bf bIHi = ldB(w2iHi, HD, nrow, kk, lane);
        v16bf bILo = ldB(w2iLo, HD, nrow, kk, lane);
        vAcc[im][in_] = wmma_bf16(aIHi, bIHi, vAcc[im][in_]);
        vAcc[im][in_] = wmma_bf16(aIHi, bILo, vAcc[im][in_]);
        vAcc[im][in_] = wmma_bf16(aILo, bIHi, vAcc[im][in_]);
      }
    }
  }

  // bias, elementwise product, LDS reduction over n (ds_add_f32)
#pragma unroll
  for (int im = 0; im < 2; im++) {
    int mloc = wave_m * 32 + im * 16 + ((lane >> 4) << 3);
#pragma unroll
    for (int j = 0; j < 8; j++) {
      float part = 0.f;
#pragma unroll
      for (int in_ = 0; in_ < 4; in_++) {
        int n = wave_n * 64 + in_ * 16 + (lane & 15);
        float uu = uAcc[im][in_][j] + b2u[n];
        float vv = vAcc[im][in_][j] + b2i[n];
        part += uu * vv;
      }
      atomicAdd(&dotLds[mloc + j], part);
    }
  }
  __syncthreads();
  if (tid < 64) out[m0 + tid] = dotLds[tid];
}

extern "C" void kernel_launch(void* const* d_in, const int* in_sizes, int n_in,
                              void* d_out, int out_size, void* d_ws,
                              size_t ws_size, hipStream_t stream) {
  (void)in_sizes; (void)n_in; (void)out_size; (void)ws_size;
  const long long* x  = (const long long*)d_in[0];
  const float* userL  = (const float*)d_in[1];
  const float* itemL  = (const float*)d_in[2];
  const float* uW1    = (const float*)d_in[3];
  const float* ub1    = (const float*)d_in[4];
  const float* uW2    = (const float*)d_in[5];
  const float* ub2    = (const float*)d_in[6];
  const float* iW1    = (const float*)d_in[7];
  const float* ib1    = (const float*)d_in[8];
  const float* iW2    = (const float*)d_in[9];
  const float* ib2    = (const float*)d_in[10];
  float* out          = (float*)d_out;

  char* ws = (char*)d_ws;
  const size_t szW1p = (size_t)HD * K2 * 2;   // 2,228,224 B per plane
  const size_t szW2p = (size_t)LD * HD * 2;   //   262,144 B per plane
  const size_t szHp  = (size_t)BD * HD * 2;   // 8,388,608 B per plane
  __bf16* w1uHi = (__bf16*)(ws + 0 * szW1p);
  __bf16* w1uLo = (__bf16*)(ws + 1 * szW1p);
  __bf16* w1iHi = (__bf16*)(ws + 2 * szW1p);
  __bf16* w1iLo = (__bf16*)(ws + 3 * szW1p);
  char* p2 = ws + 4 * szW1p;
  __bf16* w2uHi = (__bf16*)(p2 + 0 * szW2p);
  __bf16* w2uLo = (__bf16*)(p2 + 1 * szW2p);
  __bf16* w2iHi = (__bf16*)(p2 + 2 * szW2p);
  __bf16* w2iLo = (__bf16*)(p2 + 3 * szW2p);
  char* ph = p2 + 4 * szW2p;
  __bf16* hUHi = (__bf16*)(ph + 0 * szHp);
  __bf16* hULo = (__bf16*)(ph + 1 * szHp);
  __bf16* hIHi = (__bf16*)(ph + 2 * szHp);
  __bf16* hILo = (__bf16*)(ph + 3 * szHp);

  // 1) weight prep (bf16 hi/lo, permuted + transposed)
  dim3 blk(256);
  prep_w1_kernel<<<dim3((K2 * HD) / 256), blk, 0, stream>>>(uW1, w1uHi, w1uLo);
  prep_w1_kernel<<<dim3((K2 * HD) / 256), blk, 0, stream>>>(iW1, w1iHi, w1iLo);
  prep_w2_kernel<<<dim3((HD * LD) / 256), blk, 0, stream>>>(uW2, w2uHi, w2uLo);
  prep_w2_kernel<<<dim3((HD * LD) / 256), blk, 0, stream>>>(iW2, w2iHi, w2iLo);

  // 2) GEMM1, both towers in one grid: 2 x 64 m-tiles x 4 n-tiles
  gemm1_kernel<<<dim3(512), blk, 0, stream>>>(
      x, userL, itemL, w1uHi, w1uLo, w1iHi, w1iLo, ub1, ib1,
      hUHi, hULo, hIHi, hILo);

  // 3) GEMM2 + rowwise dot
  gemm2_dot_kernel<<<dim3(BD / 64), blk, 0, stream>>>(
      hUHi, hULo, hIHi, hILo, w2uHi, w2uLo, w2iHi, w2iLo, ub2, ib2, out);
}
